// EnhancedMultiHeadAttention_62397284876957
// MI455X (gfx1250) — matrix operand order repaired and checked by reading, not compiled, not run on hardware
//
#include <hip/hip_runtime.h>
#include <hip/hip_bf16.h>
#include <cstdint>
#include <cstddef>

#define DM 1024
#define SQ 2048
#define BB 4
#define NH 16
#define HD 64
#define TT (BB*SQ)

typedef _Float16 half_t;
typedef __attribute__((ext_vector_type(16))) _Float16 v16h;
typedef __attribute__((ext_vector_type(8)))  _Float16 v8h;
typedef __attribute__((ext_vector_type(4)))  _Float16 v4h;
typedef __attribute__((ext_vector_type(2)))  _Float16 v2h;
typedef __attribute__((ext_vector_type(8)))  float    v8f;

// ---------- async global->LDS copy (CDNA5 ASYNCcnt path) -----------------
#if defined(__has_builtin)
#if __has_builtin(__builtin_amdgcn_global_load_async_to_lds_b128) && \
    __has_builtin(__builtin_amdgcn_s_wait_asynccnt)
#define USE_ASYNC_LDS 1
#endif
#endif

typedef int i32x4 __attribute__((vector_size(16)));
typedef __attribute__((address_space(1))) i32x4 g_i32x4;  // global int4
typedef __attribute__((address_space(3))) i32x4 l_i32x4;  // LDS int4

static __device__ __forceinline__ void cp16(half_t* lds, const half_t* g) {
#ifdef USE_ASYNC_LDS
  __builtin_amdgcn_global_load_async_to_lds_b128((g_i32x4*)g, (l_i32x4*)lds, 0, 0);
#else
  *(v8h*)lds = *(const v8h*)g;
#endif
}

template <int N> static __device__ __forceinline__ void wait_cp() {
#ifdef USE_ASYNC_LDS
  __builtin_amdgcn_s_wait_asynccnt(N);
#endif
}

// ---------- WMMA helpers -------------------------------------------------
static __device__ __forceinline__ v8f wmma16(v16h a, v16h b, v8f c) {
  return __builtin_amdgcn_wmma_f32_16x16x32_f16(false, a, false, b, (short)0, c,
                                                false, false);
}

// A-operand fragment (16x32 f16): lane l holds row (l&15); 8-half chunks at
// kbase + (l>>4)*8 and +16 (per ISA 16-bit A layout).
static __device__ __forceinline__ v16h frag_a(const half_t* base, int row,
                                              int ldh, int kbase) {
  int l = threadIdx.x & 31;
  const half_t* p = base + (size_t)(row + (l & 15)) * ldh + kbase + ((l >> 4) << 3);
  v8h lo = *(const v8h*)p;
  v8h hi = *(const v8h*)(p + 16);
  v16h r;
#pragma unroll
  for (int i = 0; i < 8; ++i) { r[i] = lo[i]; r[i + 8] = hi[i]; }
  return r;
}

// B-operand fragment (32x16 f16): lane l holds col (l&15); 8-half chunks at
// kbase + (l>>4)*8 and +16 (matches A layout).
static __device__ __forceinline__ v16h frag_b(const half_t* base, int col,
                                              int ldh, int kbase) {
  int l = threadIdx.x & 31;
  const half_t* p = base + (size_t)(col + (l & 15)) * ldh + kbase + ((l >> 4) << 3);
  v8h lo = *(const v8h*)p;
  v8h hi = *(const v8h*)(p + 16);
  v16h r;
#pragma unroll
  for (int i = 0; i < 8; ++i) { r[i] = lo[i]; r[i + 8] = hi[i]; }
  return r;
}

// ---------- Kernel 1: weight f32 -> f16, transposed ----------------------
__global__ __launch_bounds__(256) void wtrans_kernel(const float* Wq, const float* Wk,
                                                     const float* Wv, const float* Wo,
                                                     half_t* Wt) {
  __shared__ half_t tile[32][33];
  int z = blockIdx.z;
  const float* W = (z == 0) ? Wq : (z == 1) ? Wk : (z == 2) ? Wv : Wo;
  half_t* dst = Wt + (size_t)z * DM * DM;
  int tx = threadIdx.x & 31, ty = threadIdx.x >> 5;
  int k0 = blockIdx.y * 32, n0 = blockIdx.x * 32;
#pragma unroll
  for (int j = 0; j < 4; ++j)
    tile[ty + j * 8][tx] = (half_t)W[(size_t)(k0 + ty + j * 8) * DM + n0 + tx];
  __syncthreads();
#pragma unroll
  for (int j = 0; j < 4; ++j)
    dst[(size_t)(n0 + ty + j * 8) * DM + k0 + tx] = tile[tx][ty + j * 8];
}

// ---------- block reduction ----------------------------------------------
static __device__ __forceinline__ float2 block_reduce2(float a, float b) {
  __shared__ float sA[8], sB[8];
#pragma unroll
  for (int m = 16; m >= 1; m >>= 1) {
    a += __shfl_xor(a, m, 32);
    b += __shfl_xor(b, m, 32);
  }
  if ((threadIdx.x & 31) == 0) { sA[threadIdx.x >> 5] = a; sB[threadIdx.x >> 5] = b; }
  __syncthreads();
  float ra = 0.f, rb = 0.f;
#pragma unroll
  for (int i = 0; i < 8; ++i) { ra += sA[i]; rb += sB[i]; }
  __syncthreads();
  return make_float2(ra, rb);
}

static __device__ __forceinline__ void ln_row(const float* row, const float* g,
                                              const float* bt, half_t* out,
                                              float* keep) {
  int tid = threadIdx.x;
  float4 x = *(const float4*)(row + tid * 4);
  float s  = x.x + x.y + x.z + x.w;
  float sq = x.x * x.x + x.y * x.y + x.z * x.z + x.w * x.w;
  float2 r = block_reduce2(s, sq);
  float mu  = r.x * (1.0f / DM);
  float var = r.y * (1.0f / DM) - mu * mu;
  float rs  = rsqrtf(var + 1e-5f);
  float4 gg = *(const float4*)(g + tid * 4);
  float4 b4 = *(const float4*)(bt + tid * 4);
  float y0 = (x.x - mu) * rs * gg.x + b4.x;
  float y1 = (x.y - mu) * rs * gg.y + b4.y;
  float y2 = (x.z - mu) * rs * gg.z + b4.z;
  float y3 = (x.w - mu) * rs * gg.w + b4.w;
  v4h o; o[0] = (half_t)y0; o[1] = (half_t)y1; o[2] = (half_t)y2; o[3] = (half_t)y3;
  *(v4h*)(out + tid * 4) = o;
  if (keep) { keep[0] = y0; keep[1] = y1; keep[2] = y2; keep[3] = y3; }
}

// ---------- Kernel 2: pre-norms + gate -----------------------------------
__global__ __launch_bounds__(256) void ln_gate_kernel(
    const float* q, const float* k, const float* v, const float* lnqg,
    const float* lnqb, const float* lnkvg, const float* lnkvb, const float* Wg,
    half_t* qn, half_t* kn, half_t* vn, float* gate) {
  size_t t = blockIdx.x;
  float keep[4];
  ln_row(q + t * DM, lnqg, lnqb, qn + t * DM, keep);
  ln_row(k + t * DM, lnkvg, lnkvb, kn + t * DM, nullptr);
  ln_row(v + t * DM, lnkvg, lnkvb, vn + t * DM, nullptr);
  int tid = threadIdx.x;
  float p = 0.f;
#pragma unroll
  for (int i = 0; i < 4; ++i) p += keep[i] * Wg[tid * 4 + i];
  float2 r = block_reduce2(p, 0.f);
  if (tid == 0) gate[t] = 1.0f / (1.0f + expf(-r.x));
}

// ---------- Kernel 3: QKV projections (128x128 tile WMMA GEMM) -----------
#define SKS 40    // staged row stride (halves), padded vs 64 banks
#define EPSTR 136 // epilogue tile stride (halves)
#define BUFH 10240 // halves per double-buffer stage (A 5120 + B 5120)

__global__ __launch_bounds__(256) void gemm_qkv_kernel(
    const half_t* qn, const half_t* kn, const half_t* vn, const half_t* Wt,
    const float* bq, const float* bk, const float* bv, half_t* Qo, half_t* Ko,
    half_t* VTo) {
  __shared__ half_t smem[2 * BUFH];  // epilogue reuses first 17408 halves
  int z = blockIdx.z;
  const half_t* act  = (z == 0) ? qn : (z == 1) ? kn : vn;
  const half_t* W    = Wt + (size_t)z * DM * DM;
  const float*  bias = (z == 0) ? bq : (z == 1) ? bk : bv;
  int n0 = blockIdx.x * 128, t0 = blockIdx.y * 128;
  int tid = threadIdx.x, lane = tid & 31, wave = tid >> 5;
  int wm = (wave & 3) * 32, wn = (wave >> 2) * 64;
  int hi = lane >> 4, ln = lane & 15;
  v8f acc[2][4] = {};

  auto issue = [&](int kt, int bs) {
    half_t* sA = smem + bs * BUFH;
    half_t* sB = smem + bs * BUFH + 5120;
#pragma unroll
    for (int i = 0; i < 2; ++i) {
      int c = tid + i * 256;
      int r = c >> 2, col = (c & 3) * 8;
      cp16(sA + r * SKS + col, act + (size_t)(t0 + r) * DM + kt * 32 + col);
      cp16(sB + r * SKS + col, W + (size_t)(n0 + r) * DM + kt * 32 + col);
    }
  };

  issue(0, 0);
  for (int kt = 0; kt < DM / 32; ++kt) {
    int cur = kt & 1;
    if (kt + 1 < DM / 32) {
      issue(kt + 1, cur ^ 1);
      wait_cp<4>();  // in-order: current tile's 4 copies are done
    } else {
      wait_cp<0>();
    }
    __syncthreads();  // current tile visible to all waves
    half_t* cA = smem + cur * BUFH;
    half_t* cB = smem + cur * BUFH + 5120;
    v16h af[2];
#pragma unroll
    for (int mt = 0; mt < 2; ++mt) af[mt] = frag_a(cA, wm + mt * 16, SKS, 0);
#pragma unroll
    for (int nt = 0; nt < 4; ++nt) {
      v16h bf = frag_b(cB, wn + nt * 16, SKS, 0);
#pragma unroll
      for (int mt = 0; mt < 2; ++mt) acc[mt][nt] = wmma16(af[mt], bf, acc[mt][nt]);
    }
    __syncthreads();  // all reads of buf[cur] done -> next prefetch may reuse it
  }
#pragma unroll
  for (int nt = 0; nt < 4; ++nt) {
    float bval = bias[n0 + wn + nt * 16 + ln];
#pragma unroll
    for (int mt = 0; mt < 2; ++mt)
#pragma unroll
      for (int i = 0; i < 8; ++i) acc[mt][nt][i] += bval;
  }
  if (z < 2) {  // Q/K: stage [m][n], coalesced row-major store
#pragma unroll
    for (int mt = 0; mt < 2; ++mt)
#pragma unroll
      for (int nt = 0; nt < 4; ++nt)
#pragma unroll
        for (int i = 0; i < 8; ++i)
          smem[(wm + mt * 16 + i + 8 * hi) * EPSTR + wn + nt * 16 + ln] =
              (half_t)acc[mt][nt][i];
    __syncthreads();
    half_t* out = (z == 0) ? Qo : Ko;
#pragma unroll
    for (int j = 0; j < 8; ++j) {
      int c = tid + j * 256;
      int r = c >> 4, col = (c & 15) * 8;
      *(v8h*)(out + (size_t)(t0 + r) * DM + n0 + col) =
          *(const v8h*)(smem + r * EPSTR + col);
    }
  } else {  // V: stage transposed [n][m] (packed pairs), store Vt[b][h][hd][s]
#pragma unroll
    for (int mt = 0; mt < 2; ++mt)
#pragma unroll
      for (int nt = 0; nt < 4; ++nt)
#pragma unroll
        for (int i = 0; i < 8; i += 2) {
          v2h pk; pk[0] = (half_t)acc[mt][nt][i]; pk[1] = (half_t)acc[mt][nt][i + 1];
          *(v2h*)(smem + (size_t)(wn + nt * 16 + ln) * EPSTR + wm + mt * 16 + i +
                  8 * hi) = pk;
        }
    __syncthreads();
    int bidx = t0 >> 11;
    int s0 = t0 & (SQ - 1);
#pragma unroll
    for (int j = 0; j < 8; ++j) {
      int c = tid + j * 256;
      int r = c >> 4, col = (c & 15) * 8;
      int ng = n0 + r;
      int h = ng >> 6, hd = ng & 63;
      *(v8h*)(VTo + ((size_t)((bidx * NH + h) * HD + hd)) * SQ + s0 + col) =
          *(const v8h*)(smem + r * EPSTR + col);
    }
  }
}

// ---------- Kernel 4: flash attention ------------------------------------
#define AST 72  // attention LDS row stride (halves)

__global__ __launch_bounds__(256) void attn_kernel(const half_t* Q, const half_t* K,
                                                   const half_t* VT, half_t* AO) {
  __shared__ half_t sK[2][64 * AST];
  __shared__ half_t sV[2][64 * AST];
  __shared__ half_t sP[8 * 16 * AST];
  int tid = threadIdx.x, lane = tid & 31, wave = tid >> 5;
  int hi = lane >> 4, ln = lane & 15;
  int bh = blockIdx.y;
  int b = bh >> 4, h = bh & 15;
  int qr = blockIdx.x * 128 + wave * 16;
  half_t* myP = sP + wave * 16 * AST;
  const float L2E = 1.44269504f;

  const half_t* qbase = Q + (size_t)(b * SQ + qr) * DM + h * 64;
  v16h qf[2];
  qf[0] = frag_a(qbase, 0, DM, 0);
  qf[1] = frag_a(qbase, 0, DM, 32);

  float mrow[8], lrow[8];
#pragma unroll
  for (int i = 0; i < 8; ++i) { mrow[i] = -1e30f; lrow[i] = 0.f; }
  v8f acc[4] = {};

  auto issue = [&](int kt, int bs) {
#pragma unroll
    for (int i = 0; i < 2; ++i) {
      int c = tid + i * 256;
      int r = c >> 3, col = (c & 7) * 8;
      cp16(&sK[bs][r * AST + col],
           K + (size_t)(b * SQ + kt * 64 + r) * DM + h * 64 + col);
      cp16(&sV[bs][r * AST + col],
           VT + (size_t)((b * NH + h) * HD + r) * SQ + kt * 64 + col);
    }
  };

  issue(0, 0);
  for (int kt = 0; kt < SQ / 64; ++kt) {
    int cur = kt & 1;
    if (kt + 1 < SQ / 64) {
      issue(kt + 1, cur ^ 1);
      wait_cp<4>();
    } else {
      wait_cp<0>();
    }
    __syncthreads();

    v8f sc[4] = {};
#pragma unroll
    for (int nt = 0; nt < 4; ++nt)
#pragma unroll
      for (int ks = 0; ks < 2; ++ks) {
        v16h bf = frag_b(sK[cur], nt * 16, AST, ks * 32);
        sc[nt] = wmma16(qf[ks], bf, sc[nt]);
      }
#pragma unroll
    for (int nt = 0; nt < 4; ++nt)
#pragma unroll
      for (int i = 0; i < 8; ++i) sc[nt][i] *= 0.125f;  // Hd^-0.5

    // online softmax: rows split lanes 0-15 (m=i) / 16-31 (m=i+8)
    float rm[8];
#pragma unroll
    for (int i = 0; i < 8; ++i)
      rm[i] = fmaxf(fmaxf(sc[0][i], sc[1][i]), fmaxf(sc[2][i], sc[3][i]));
#pragma unroll
    for (int m = 8; m >= 1; m >>= 1)
#pragma unroll
      for (int i = 0; i < 8; ++i) rm[i] = fmaxf(rm[i], __shfl_xor(rm[i], m, 32));

    float nm[8], corr[8], rs[8];
#pragma unroll
    for (int i = 0; i < 8; ++i) {
      nm[i] = fmaxf(mrow[i], rm[i]);
      corr[i] = exp2f((mrow[i] - nm[i]) * L2E);
      rs[i] = 0.f;
    }
#pragma unroll
    for (int nt = 0; nt < 4; ++nt)
#pragma unroll
      for (int i = 0; i < 8; ++i) {
        float p = exp2f((sc[nt][i] - nm[i]) * L2E);
        rs[i] += p;
        myP[(i + 8 * hi) * AST + nt * 16 + ln] = (half_t)p;
      }
#pragma unroll
    for (int m = 8; m >= 1; m >>= 1)
#pragma unroll
      for (int i = 0; i < 8; ++i) rs[i] += __shfl_xor(rs[i], m, 32);
#pragma unroll
    for (int i = 0; i < 8; ++i) {
      lrow[i] = lrow[i] * corr[i] + rs[i];
      mrow[i] = nm[i];
    }
#pragma unroll
    for (int dt = 0; dt < 4; ++dt)
#pragma unroll
      for (int i = 0; i < 8; ++i) acc[dt][i] *= corr[i];
    __syncthreads();  // P visible (and keeps barrier counts uniform)

#pragma unroll
    for (int ks = 0; ks < 2; ++ks) {
      v16h pf = frag_a(myP, 0, AST, ks * 32);
#pragma unroll
      for (int dt = 0; dt < 4; ++dt) {
        v16h bf = frag_b(sV[cur], dt * 16, AST, ks * 32);
        acc[dt] = wmma16(pf, bf, acc[dt]);
      }
    }
    __syncthreads();  // all reads of buf[cur] done -> prefetch may reuse it
  }

  float inv[8];
#pragma unroll
  for (int i = 0; i < 8; ++i) inv[i] = 1.0f / lrow[i];
#pragma unroll
  for (int dt = 0; dt < 4; ++dt)
#pragma unroll
    for (int i = 0; i < 8; ++i)
      myP[(i + 8 * hi) * AST + dt * 16 + ln] = (half_t)(acc[dt][i] * inv[i]);
  __syncthreads();
#pragma unroll
  for (int j = 0; j < 4; ++j) {
    int c = lane + j * 32;
    int r = c >> 3, col = (c & 7) * 8;
    *(v8h*)(AO + (size_t)(b * SQ + qr + r) * DM + h * 64 + col) =
        *(const v8h*)(myP + r * AST + col);
  }
}

// ---------- Kernel 5: output projection + gate ---------------------------
__global__ __launch_bounds__(256) void oproj_kernel(const half_t* AOin,
                                                    const half_t* Wot,
                                                    const float* bo,
                                                    const float* gate, float* tmp) {
  __shared__ half_t smem[2 * BUFH];
  __shared__ float sg[128];
  int n0 = blockIdx.x * 128, t0 = blockIdx.y * 128;
  int tid = threadIdx.x, lane = tid & 31, wave = tid >> 5;
  int wm = (wave & 3) * 32, wn = (wave >> 2) * 64;
  int hi = lane >> 4, ln = lane & 15;
  if (tid < 128) sg[tid] = gate[t0 + tid];
  v8f acc[2][4] = {};

  auto issue = [&](int kt, int bs) {
    half_t* sA = smem + bs * BUFH;
    half_t* sB = smem + bs * BUFH + 5120;
#pragma unroll
    for (int i = 0; i < 2; ++i) {
      int c = tid + i * 256;
      int r = c >> 2, col = (c & 3) * 8;
      cp16(sA + r * SKS + col, AOin + (size_t)(t0 + r) * DM + kt * 32 + col);
      cp16(sB + r * SKS + col, Wot + (size_t)(n0 + r) * DM + kt * 32 + col);
    }
  };

  issue(0, 0);
  for (int kt = 0; kt < DM / 32; ++kt) {
    int cur = kt & 1;
    if (kt + 1 < DM / 32) {
      issue(kt + 1, cur ^ 1);
      wait_cp<4>();
    } else {
      wait_cp<0>();
    }
    __syncthreads();
    half_t* cA = smem + cur * BUFH;
    half_t* cB = smem + cur * BUFH + 5120;
    v16h af[2];
#pragma unroll
    for (int mt = 0; mt < 2; ++mt) af[mt] = frag_a(cA, wm + mt * 16, SKS, 0);
#pragma unroll
    for (int nt = 0; nt < 4; ++nt) {
      v16h bf = frag_b(cB, wn + nt * 16, SKS, 0);
#pragma unroll
      for (int mt = 0; mt < 2; ++mt) acc[mt][nt] = wmma16(af[mt], bf, acc[mt][nt]);
    }
    __syncthreads();
  }
#pragma unroll
  for (int nt = 0; nt < 4; ++nt) {
    int col = n0 + wn + nt * 16 + ln;
    float bval = bo[col];
#pragma unroll
    for (int mt = 0; mt < 2; ++mt)
#pragma unroll
      for (int i = 0; i < 8; ++i) {
        int row = wm + mt * 16 + i + 8 * hi;
        tmp[(size_t)(t0 + row) * DM + col] = (acc[mt][nt][i] + bval) * sg[row];
      }
  }
}

// ---------- Kernel 6: final LayerNorm ------------------------------------
__global__ __launch_bounds__(256) void lnout_kernel(const float* tmp, const float* g,
                                                    const float* bt, float* out) {
  size_t t = blockIdx.x;
  int tid = threadIdx.x;
  float4 x = *(const float4*)(tmp + t * DM + tid * 4);
  float s  = x.x + x.y + x.z + x.w;
  float sq = x.x * x.x + x.y * x.y + x.z * x.z + x.w * x.w;
  float2 r = block_reduce2(s, sq);
  float mu  = r.x * (1.0f / DM);
  float var = r.y * (1.0f / DM) - mu * mu;
  float rs  = rsqrtf(var + 1e-5f);
  float4 gg = *(const float4*)(g + tid * 4);
  float4 b4 = *(const float4*)(bt + tid * 4);
  float4 y;
  y.x = (x.x - mu) * rs * gg.x + b4.x;
  y.y = (x.y - mu) * rs * gg.y + b4.y;
  y.z = (x.z - mu) * rs * gg.z + b4.z;
  y.w = (x.w - mu) * rs * gg.w + b4.w;
  *(float4*)(out + t * DM + tid * 4) = y;
}

// ---------- host ----------------------------------------------------------
extern "C" void kernel_launch(void* const* d_in, const int* in_sizes, int n_in,
                              void* d_out, int out_size, void* d_ws, size_t ws_size,
                              hipStream_t stream) {
  const float* query = (const float*)d_in[0];
  const float* key   = (const float*)d_in[1];
  const float* value = (const float*)d_in[2];
  const float* Wq    = (const float*)d_in[3];
  const float* bq    = (const float*)d_in[4];
  const float* Wk    = (const float*)d_in[5];
  const float* bk    = (const float*)d_in[6];
  const float* Wv    = (const float*)d_in[7];
  const float* bv    = (const float*)d_in[8];
  const float* Wo    = (const float*)d_in[9];
  const float* bo    = (const float*)d_in[10];
  const float* Wg    = (const float*)d_in[11];
  const float* lnqg  = (const float*)d_in[12];
  const float* lnqb  = (const float*)d_in[13];
  const float* lnkvg = (const float*)d_in[14];
  const float* lnkvb = (const float*)d_in[15];
  const float* lnog  = (const float*)d_in[16];
  const float* lnob  = (const float*)d_in[17];

  const size_t MB = (size_t)1 << 20;
  char* ws = (char*)d_ws;
  half_t* Wt   = (half_t*)(ws + 0 * MB);    // 4 x DxD f16 (transposed): 8 MB
  half_t* qn   = (half_t*)(ws + 8 * MB);    // 16 MB (reused as attention out)
  half_t* kn   = (half_t*)(ws + 24 * MB);   // 16 MB
  half_t* vn   = (half_t*)(ws + 40 * MB);   // 16 MB
  float*  tmp  = (float*)(ws + 24 * MB);    // 32 MB, aliases kn/vn (dead then)
  half_t* Qb   = (half_t*)(ws + 56 * MB);   // 16 MB
  half_t* Kb   = (half_t*)(ws + 72 * MB);   // 16 MB
  half_t* VTb  = (half_t*)(ws + 88 * MB);   // 16 MB, V transposed [B,H,Hd,S]
  float*  gate = (float*)(ws + 104 * MB);   // 32 KB
  half_t* AO   = qn;                        // attention output (qn is dead)

  wtrans_kernel<<<dim3(DM / 32, DM / 32, 4), 256, 0, stream>>>(Wq, Wk, Wv, Wo, Wt);
  ln_gate_kernel<<<TT, 256, 0, stream>>>(query, key, value, lnqg, lnqb, lnkvg,
                                         lnkvb, Wg, qn, kn, vn, gate);
  gemm_qkv_kernel<<<dim3(DM / 128, TT / 128, 3), 256, 0, stream>>>(
      qn, kn, vn, Wt, bq, bk, bv, Qb, Kb, VTb);
  attn_kernel<<<dim3(SQ / 128, BB * NH), 256, 0, stream>>>(Qb, Kb, VTb, AO);
  oproj_kernel<<<dim3(DM / 128, TT / 128), 256, 0, stream>>>(
      AO, Wt + (size_t)3 * DM * DM, bo, gate, tmp);
  lnout_kernel<<<TT, 256, 0, stream>>>(tmp, lnog, lnob, (float*)d_out);
}
